// MultiHeadAttention_85907935854652
// MI455X (gfx1250) — matrix operand order repaired and checked by reading, not compiled, not run on hardware
//
#include <hip/hip_runtime.h>

// ---------------------------------------------------------------------------
// MHA forward for MI455X (gfx1250, wave32, WMMA 16x16x32 f16 -> f32).
// HBM-bound on the 537MB attn tensor; all reusable operands staged as f16,
// and ALL WMMA B-operands are pre-transposed so every lane reads two
// contiguous 16B runs per 32x16 tile.
// ---------------------------------------------------------------------------

typedef __attribute__((ext_vector_type(16))) _Float16 v16h;
typedef __attribute__((ext_vector_type(8)))  float    v8f;

#define S_LEN 2048
#define D_MODEL 1024
#define N_HEAD 16
#define ATT_DIM 64
#define BATCH 2

// ---- WMMA wrapper ---------------------------------------------------------
__device__ __forceinline__ v8f wmma16(v16h a, v16h b, v8f c) {
  // D = A(16x32 f16) * B(32x16 f16) + C(16x16 f32)
  return __builtin_amdgcn_wmma_f32_16x16x32_f16(
      /*neg_a=*/false, a, /*neg_b=*/false, b,
      /*c_mod=*/(short)0, c, /*reuse_a=*/false, /*reuse_b=*/false);
}

__device__ __forceinline__ void zero8(v8f& c) {
#pragma unroll
  for (int i = 0; i < 8; ++i) c[i] = 0.0f;
}

// ---- A-operand loaders (16x32 tile, ISA layout) ---------------------------
// lanes 0-15: row M=lane,   elems 0..7 = K0..7,  elems 8..15 = K16..23
// lanes16-31: row M=lane-16, elems 0..7 = K8..15, elems 8..15 = K24..31
__device__ __forceinline__ v16h load_A_f32(const float* __restrict__ A, int lda,
                                           int m0, int k0) {
  const int lane = threadIdx.x & 31;
  const int row  = m0 + (lane & 15);
  const int kb   = (lane >> 4) * 8;
  const float* p = A + (size_t)row * lda + k0 + kb;
  v16h a;
#pragma unroll
  for (int j = 0; j < 8; ++j) a[j] = (_Float16)p[j];
#pragma unroll
  for (int j = 0; j < 8; ++j) a[8 + j] = (_Float16)p[16 + j];
  return a;
}

__device__ __forceinline__ v16h load_A_f16(const _Float16* __restrict__ A,
                                           int lda, int m0, int k0) {
  const int lane = threadIdx.x & 31;
  const int row  = m0 + (lane & 15);
  const int kb   = (lane >> 4) * 8;
  const _Float16* p = A + (size_t)row * lda + k0 + kb;
  v16h a;
#pragma unroll
  for (int j = 0; j < 8; ++j) a[j] = p[j];
#pragma unroll
  for (int j = 0; j < 8; ++j) a[8 + j] = p[16 + j];
  return a;
}

// ---- B-operand loader (32x16 tile, ISA layout) ----------------------------
// lane: col N = lane&15 ; lanes 0-15 hold K=0..7,16..23, lanes 16-31 hold K=8..15,24..31
// Transposed NxK f16 source (contiguous along K -> two 16B runs per lane):
__device__ __forceinline__ v16h load_B_tr_f16(const _Float16* __restrict__ Bt,
                                              int ldt, int k0, int n0) {
  const int lane = threadIdx.x & 31;
  const int n    = n0 + (lane & 15);
  const int kb   = (lane >> 4) * 8;
  const _Float16* p = Bt + (size_t)n * ldt + k0 + kb;
  v16h b;
#pragma unroll
  for (int j = 0; j < 8; ++j) b[j] = p[j];
#pragma unroll
  for (int j = 0; j < 8; ++j) b[8 + j] = p[16 + j];
  return b;
}

// ---- C/D stores (16x16 f32 accumulator, ISA layout) -----------------------
// lane: col N = lane&15 ; vgpr i -> row M = i + 8*(lane>=16)
__device__ __forceinline__ void store_C_f32(float* __restrict__ C, int ldc,
                                            int m0, int n0, v8f c, float scale) {
  const int lane = threadIdx.x & 31;
  const int n    = n0 + (lane & 15);
  const int mb   = m0 + (lane >> 4) * 8;
#pragma unroll
  for (int i = 0; i < 8; ++i) C[(size_t)(mb + i) * ldc + n] = c[i] * scale;
}

__device__ __forceinline__ void store_C_f16(_Float16* __restrict__ C, int ldc,
                                            int m0, int n0, v8f c) {
  const int lane = threadIdx.x & 31;
  const int n    = n0 + (lane & 15);
  const int mb   = m0 + (lane >> 4) * 8;
#pragma unroll
  for (int i = 0; i < 8; ++i) C[(size_t)(mb + i) * ldc + n] = (_Float16)c[i];
}

// transposed store: element (m,n) -> C[n*ldt + m] (contiguous in vgpr index)
__device__ __forceinline__ void store_C_f16_tr(_Float16* __restrict__ C, int ldt,
                                               int m0, int n0, v8f c) {
  const int lane = threadIdx.x & 31;
  const int n    = n0 + (lane & 15);
  const int mb   = m0 + (lane >> 4) * 8;
  _Float16* p = C + (size_t)n * ldt + mb;
#pragma unroll
  for (int i = 0; i < 8; ++i) p[i] = (_Float16)c[i];
}

// ---------------------------------------------------------------------------
// Kernel 0a: transpose QKV weights [h][d][e] f32 -> [h][e][d] f16 (2MB each,
// L2 resident, reused by 1024 projection blocks).  grid=(4096,3), block=256.
// ---------------------------------------------------------------------------
__global__ void k_wqkv_t(const float* __restrict__ Wq, const float* __restrict__ Wk,
                         const float* __restrict__ Wv, _Float16* __restrict__ wqt,
                         _Float16* __restrict__ wkt, _Float16* __restrict__ wvt) {
  const int idx = blockIdx.x * 256 + threadIdx.x;  // 0 .. H*D*E-1 (=1M)
  const int zz  = blockIdx.y;
  const float* W  = (zz == 0 ? Wq : zz == 1 ? Wk : Wv);
  _Float16*    Wt = (zz == 0 ? wqt : zz == 1 ? wkt : wvt);
  const int h = idx >> 16;            // / (D*E)
  const int r = idx & 0xFFFF;
  const int d = r >> 6;               // / E
  const int e = r & (ATT_DIM - 1);
  Wt[(size_t)h * ATT_DIM * D_MODEL + (size_t)e * D_MODEL + d] = (_Float16)W[idx];
}

// ---------------------------------------------------------------------------
// Kernel 0b: transpose Wo [k][n] f32 -> Wo^T [n][k] f16 (2MB, L2 resident)
// ---------------------------------------------------------------------------
__global__ void k_wo_t(const float* __restrict__ Wo, _Float16* __restrict__ wot) {
  const int idx = blockIdx.x * 256 + threadIdx.x;  // exactly 1M threads
  const int n  = idx & (D_MODEL - 1);
  const int kk = idx >> 10;
  wot[(size_t)n * D_MODEL + kk] = (_Float16)Wo[idx];
}

// ---------------------------------------------------------------------------
// Kernel 1: per-head QKV projection.  grid = (S/64, H*B, 3), block = 128.
// z=0: qh[h,b,s,e] f16   z=1: kh[h,b,s,e] f16   z=2: vh^T[h,b,e,s] f16
// Each wave: 16(M) x 64(N) strip, K = 1024 in steps of 32; B from f16
// transposed weights (contiguous reads).
// ---------------------------------------------------------------------------
__global__ void k_project(const float* __restrict__ q, const float* __restrict__ k,
                          const float* __restrict__ v,
                          const _Float16* __restrict__ wqt,
                          const _Float16* __restrict__ wkt,
                          const _Float16* __restrict__ wvt,
                          _Float16* __restrict__ qh, _Float16* __restrict__ kh,
                          _Float16* __restrict__ vht) {
  const int wave = threadIdx.x >> 5;
  const int hb   = blockIdx.y;
  const int h    = hb >> 1;
  const int b    = hb & 1;
  const int s0   = (blockIdx.x * 4 + wave) * 16;
  const int zz   = blockIdx.z;

  const float* X = (zz == 0 ? q : zz == 1 ? k : v) + (size_t)b * S_LEN * D_MODEL;
  const _Float16* Wt =
      (zz == 0 ? wqt : zz == 1 ? wkt : wvt) + (size_t)h * ATT_DIM * D_MODEL;

  v8f acc[4];
#pragma unroll
  for (int t = 0; t < 4; ++t) zero8(acc[t]);

  for (int k0 = 0; k0 < D_MODEL; k0 += 32) {
    v16h a = load_A_f32(X, D_MODEL, s0, k0);
#pragma unroll
    for (int t = 0; t < 4; ++t) {
      // B(k=d, n=e) = Wt[e*D + d] -> transposed source, contiguous in K
      v16h bt = load_B_tr_f16(Wt, D_MODEL, k0, t * 16);
      acc[t] = wmma16(a, bt, acc[t]);
    }
  }

  if (zz < 2) {
    _Float16* out = (zz == 0 ? qh : kh) + (size_t)hb * S_LEN * ATT_DIM;
#pragma unroll
    for (int t = 0; t < 4; ++t) store_C_f16(out, ATT_DIM, s0, t * 16, acc[t]);
  } else {
    _Float16* out = vht + (size_t)hb * ATT_DIM * S_LEN;
#pragma unroll
    for (int t = 0; t < 4; ++t) store_C_f16_tr(out, S_LEN, s0, t * 16, acc[t]);
  }
}

// ---------------------------------------------------------------------------
// Kernel 2: scores = qh * kh^T / 64, written f32 straight into attn output.
// grid = (S/64, S/64, H*B), block = 128.  M=N=2048, K=64 per (h,b).
// ---------------------------------------------------------------------------
__global__ void k_scores(const _Float16* __restrict__ qh,
                         const _Float16* __restrict__ kh,
                         float* __restrict__ attn) {
  const int wave = threadIdx.x >> 5;
  const int hb   = blockIdx.z;
  const int s0   = (blockIdx.y * 4 + wave) * 16;
  const int n0b  = blockIdx.x * 64;

  const _Float16* Q = qh + (size_t)hb * S_LEN * ATT_DIM;
  const _Float16* K = kh + (size_t)hb * S_LEN * ATT_DIM;

  v8f acc[4];
#pragma unroll
  for (int t = 0; t < 4; ++t) zero8(acc[t]);

#pragma unroll
  for (int k0 = 0; k0 < ATT_DIM; k0 += 32) {
    v16h a = load_A_f16(Q, ATT_DIM, s0, k0);
#pragma unroll
    for (int t = 0; t < 4; ++t) {
      // B(k=e, n=k_idx) = kh[k_idx*64 + e] -> kh is the transposed source
      v16h bt = load_B_tr_f16(K, ATT_DIM, k0, n0b + t * 16);
      acc[t] = wmma16(a, bt, acc[t]);
    }
  }

  float* out = attn + (size_t)hb * S_LEN * S_LEN;
#pragma unroll
  for (int t = 0; t < 4; ++t)
    store_C_f32(out, S_LEN, s0, n0b + t * 16, acc[t], 1.0f / (float)ATT_DIM);
}

// ---------------------------------------------------------------------------
// Kernel 3: in-place row softmax over attn. 1 block (256 thr) per row of 2048.
// Row lives in registers (8 vals/thread); shfl_xor wave32 + LDS reductions.
// ---------------------------------------------------------------------------
__global__ void k_softmax(float* __restrict__ attn) {
  float* p = attn + (size_t)blockIdx.x * S_LEN;
  const int t    = threadIdx.x;
  const int lane = t & 31;
  const int wid  = t >> 5;

  float v[8];
  float m = -3.4e38f;
#pragma unroll
  for (int i = 0; i < 8; ++i) {
    v[i] = p[t * 8 + i];
    m = fmaxf(m, v[i]);
  }
#pragma unroll
  for (int o = 16; o > 0; o >>= 1) m = fmaxf(m, __shfl_xor(m, o, 32));

  __shared__ float smax[8];
  __shared__ float ssum[8];
  if (lane == 0) smax[wid] = m;
  __syncthreads();
  m = smax[0];
#pragma unroll
  for (int j = 1; j < 8; ++j) m = fmaxf(m, smax[j]);

  float s = 0.0f;
#pragma unroll
  for (int i = 0; i < 8; ++i) {
    v[i] = __expf(v[i] - m);
    s += v[i];
  }
#pragma unroll
  for (int o = 16; o > 0; o >>= 1) s += __shfl_xor(s, o, 32);
  if (lane == 0) ssum[wid] = s;
  __syncthreads();
  s = 0.0f;
#pragma unroll
  for (int j = 0; j < 8; ++j) s += ssum[j];

  const float inv = 1.0f / s;
#pragma unroll
  for (int i = 0; i < 8; ++i) p[t * 8 + i] = v[i] * inv;
}

// ---------------------------------------------------------------------------
// Kernel 4: att = attn * vh, stored directly in concat layout cat[b,s,h*64+e].
// grid = (S/64, H*B), block = 128.  M=2048, N=64, K=2048 per (h,b).
// attn read f32 -> f16 on the fly (A operand); vh^T f16 (contiguous B reads).
// ---------------------------------------------------------------------------
__global__ void k_att(const float* __restrict__ attn,
                      const _Float16* __restrict__ vht,
                      float* __restrict__ cat) {
  const int wave = threadIdx.x >> 5;
  const int hb   = blockIdx.y;
  const int h    = hb >> 1;
  const int b    = hb & 1;
  const int s0   = (blockIdx.x * 4 + wave) * 16;

  const float* A     = attn + (size_t)hb * S_LEN * S_LEN;
  const _Float16* Vt = vht + (size_t)hb * ATT_DIM * S_LEN;

  v8f acc[4];
#pragma unroll
  for (int t = 0; t < 4; ++t) zero8(acc[t]);

  for (int k0 = 0; k0 < S_LEN; k0 += 32) {
    __builtin_prefetch(&A[(size_t)(s0 + (threadIdx.x & 15)) * S_LEN + k0 + 64], 0, 1);
    v16h a = load_A_f32(A, S_LEN, s0, k0);
#pragma unroll
    for (int t = 0; t < 4; ++t) {
      v16h bt = load_B_tr_f16(Vt, S_LEN, k0, t * 16);
      acc[t] = wmma16(a, bt, acc[t]);
    }
  }

  float* out = cat + (size_t)b * S_LEN * D_MODEL + h * ATT_DIM;
#pragma unroll
  for (int t = 0; t < 4; ++t)
    store_C_f32(out, D_MODEL, s0, t * 16, acc[t], 1.0f);
}

// ---------------------------------------------------------------------------
// Kernel 5: out = cat @ Wo.  M=4096, N=1024, K=1024.
// grid = (1024/64, 4096/64), block = 128.
// ---------------------------------------------------------------------------
__global__ void k_out(const float* __restrict__ cat,
                      const _Float16* __restrict__ wot,
                      float* __restrict__ out) {
  const int wave = threadIdx.x >> 5;
  const int m0   = (blockIdx.y * 4 + wave) * 16;
  const int n0b  = blockIdx.x * 64;

  v8f acc[4];
#pragma unroll
  for (int t = 0; t < 4; ++t) zero8(acc[t]);

  for (int k0 = 0; k0 < D_MODEL; k0 += 32) {
    __builtin_prefetch(&cat[(size_t)(m0 + (threadIdx.x & 15)) * D_MODEL + k0 + 64], 0, 1);
    v16h a = load_A_f32(cat, D_MODEL, m0, k0);
#pragma unroll
    for (int t = 0; t < 4; ++t) {
      v16h bt = load_B_tr_f16(wot, D_MODEL, k0, n0b + t * 16);
      acc[t] = wmma16(a, bt, acc[t]);
    }
  }

#pragma unroll
  for (int t = 0; t < 4; ++t)
    store_C_f32(out, D_MODEL, m0, n0b + t * 16, acc[t], 1.0f);
}

// ---------------------------------------------------------------------------
extern "C" void kernel_launch(void* const* d_in, const int* in_sizes, int n_in,
                              void* d_out, int out_size, void* d_ws, size_t ws_size,
                              hipStream_t stream) {
  (void)in_sizes; (void)n_in; (void)out_size; (void)ws_size;

  const float* q  = (const float*)d_in[0];
  const float* k  = (const float*)d_in[1];
  const float* v  = (const float*)d_in[2];
  const float* Wq = (const float*)d_in[3];
  const float* Wk = (const float*)d_in[4];
  const float* Wv = (const float*)d_in[5];
  const float* Wo = (const float*)d_in[6];

  float* outp = (float*)d_out;                                     // [B,S,D]
  float* attn = outp + (size_t)BATCH * S_LEN * D_MODEL;            // [H,B,S,S]

  // Workspace partition (~49 MB)
  char* w = (char*)d_ws;
  const size_t proj_bytes = (size_t)N_HEAD * BATCH * S_LEN * ATT_DIM * sizeof(_Float16);
  const size_t wt_bytes   = (size_t)N_HEAD * ATT_DIM * D_MODEL * sizeof(_Float16);
  _Float16* qh  = (_Float16*)w; w += proj_bytes;                   // [h,b,s,e]
  _Float16* kh  = (_Float16*)w; w += proj_bytes;                   // [h,b,s,e]
  _Float16* vht = (_Float16*)w; w += proj_bytes;                   // [h,b,e,s]
  float* cat    = (float*)w;    w += (size_t)BATCH * S_LEN * D_MODEL * sizeof(float);
  _Float16* wot = (_Float16*)w; w += (size_t)D_MODEL * D_MODEL * sizeof(_Float16);
  _Float16* wqt = (_Float16*)w; w += wt_bytes;                     // [h,e,d]
  _Float16* wkt = (_Float16*)w; w += wt_bytes;                     // [h,e,d]
  _Float16* wvt = (_Float16*)w;                                    // [h,e,d]

  // 0) weight transposes (f32 -> f16, B-operand friendly layouts)
  k_wqkv_t<<<dim3((N_HEAD * D_MODEL * ATT_DIM) / 256, 3), 256, 0, stream>>>(
      Wq, Wk, Wv, wqt, wkt, wvt);
  k_wo_t<<<dim3((D_MODEL * D_MODEL) / 256), 256, 0, stream>>>(Wo, wot);

  // 1) QKV projections (z selects q/k/v)
  k_project<<<dim3(S_LEN / 64, N_HEAD * BATCH, 3), 128, 0, stream>>>(
      q, k, v, wqt, wkt, wvt, qh, kh, vht);

  // 2) scores -> attn region of d_out
  k_scores<<<dim3(S_LEN / 64, S_LEN / 64, N_HEAD * BATCH), 128, 0, stream>>>(
      qh, kh, attn);

  // 3) in-place softmax, one block per row
  k_softmax<<<dim3(N_HEAD * BATCH * S_LEN), 256, 0, stream>>>(attn);

  // 4) attn @ vh -> concat layout
  k_att<<<dim3(S_LEN / 64, N_HEAD * BATCH), 128, 0, stream>>>(attn, vht, cat);

  // 5) cat @ Wo -> out
  k_out<<<dim3(D_MODEL / 64, (BATCH * S_LEN) / 64), 128, 0, stream>>>(cat, wot, outp);
}
